// Mamba2PoSTBlock_42125039239465
// MI455X (gfx1250) — matrix operand order repaired and checked by reading, not compiled, not run on hardware
//
#include <hip/hip_runtime.h>
#include <stdint.h>

// ---------------- geometry ----------------
#define D_MODEL    1024
#define D_INNER    2048
#define HEADDIM    64
#define NHEADS     32
#define D_STATE    128
#define D_CONV     4
#define D_CONV_CH  2304      // D_INNER + 2*D_STATE
#define D_IN_PROJ  4384      // 2*D_INNER + 2*D_STATE + NHEADS
#define ZXS        4480      // D_IN_PROJ padded to a multiple of 128 (35*128)
#define CHUNK      256
#define BATCH      2
#define SEQLEN     4096
#define NCHUNK     (SEQLEN / CHUNK)      // 16
#define ROWS       (BATCH * SEQLEN)      // 8192
#define EPS        1e-5f

typedef __attribute__((ext_vector_type(16))) __bf16 v16bf;
typedef __attribute__((ext_vector_type(8)))  float  v8f;
typedef __attribute__((ext_vector_type(4)))  unsigned v4u;
typedef __attribute__((ext_vector_type(8)))  int      v8i;
typedef __attribute__((ext_vector_type(4)))  int      v4i;

// Tensor Data Mover availability (this toolchain: 6-arg clang-23 form)
#ifndef USE_TDM
#define USE_TDM 1
#endif
#if USE_TDM && defined(__has_builtin)
#  if __has_builtin(__builtin_amdgcn_tensor_load_to_lds) && __has_builtin(__builtin_amdgcn_s_wait_tensorcnt)
#    define HAVE_TDM 1
#  endif
#endif
#ifndef HAVE_TDM
#define HAVE_TDM 0
#endif

// ---------------- helpers ----------------
__device__ __forceinline__ unsigned short f2bf(float f) {
  union { float f; unsigned int u; } c; c.f = f;
  unsigned int u = c.u;
  unsigned int r = u + 0x7fffu + ((u >> 16) & 1u);   // round-to-nearest-even
  return (unsigned short)(r >> 16);
}
__device__ __forceinline__ float bf2f(unsigned short h) {
  union { unsigned int u; float f; } c; c.u = ((unsigned int)h) << 16;
  return c.f;
}
__device__ __forceinline__ float sigmoidf_(float x) { return 1.0f / (1.0f + __expf(-x)); }

// Load a v16bf fragment as two contiguous 16B runs (two ds_load_b128).
// CDNA5 16-bit A-operand layout == {k in [8h,8h+8)} ++ {k in [16+8h,16+8h+8)};
// B-operand layout (lane=col) == same k order as A -> also two 16B runs.
__device__ __forceinline__ v16bf ldfrag(const unsigned short* p0, const unsigned short* p1) {
  union { uint4 q[2]; v16bf v; } u;
  u.q[0] = *(const uint4*)p0;
  u.q[1] = *(const uint4*)p1;
  return u.v;
}

// ---------------- f32 -> bf16 conversion ----------------
__global__ void f2bf_kernel(const float* __restrict__ in,
                            unsigned short* __restrict__ out, long long n) {
  long long i = (long long)blockIdx.x * 256 + threadIdx.x;
  if (i < n) out[i] = f2bf(in[i]);
}

// f32 [rows x ncols] -> bf16 [rows x npad], zero-padded columns
__global__ void f2bf_pad_kernel(const float* __restrict__ in,
                                unsigned short* __restrict__ out,
                                int ncols, int npad, long long total) {
  long long i = (long long)blockIdx.x * 256 + threadIdx.x;
  if (i >= total) return;
  int col = (int)(i % npad);
  long long row = i / npad;
  out[i] = (col < ncols) ? f2bf(in[row * ncols + col]) : (unsigned short)0;
}

// ---------------- residual add + RMSNorm -> bf16 ----------------
__global__ __launch_bounds__(256) void addnorm_kernel(
    const float* __restrict__ hid, const float* __restrict__ res,
    const float* __restrict__ w, float* __restrict__ res_out,
    unsigned short* __restrict__ h_bf)
{
  int row = blockIdx.x;
  const float* hp = hid + (size_t)row * D_MODEL;
  const float* rp = res + (size_t)row * D_MODEL;
  float*       ro = res_out + (size_t)row * D_MODEL;
  unsigned short* ho = h_bf + (size_t)row * D_MODEL;
  __shared__ float red[256];
  float v[4]; float ss = 0.f;
  #pragma unroll
  for (int j = 0; j < 4; ++j) {
    int i = threadIdx.x + j * 256;
    float x = hp[i] + rp[i];
    v[j] = x; ro[i] = x; ss += x * x;
  }
  red[threadIdx.x] = ss; __syncthreads();
  for (int o = 128; o > 0; o >>= 1) {
    if ((int)threadIdx.x < o) red[threadIdx.x] += red[threadIdx.x + o];
    __syncthreads();
  }
  float scale = rsqrtf(red[0] / (float)D_MODEL + EPS);
  #pragma unroll
  for (int j = 0; j < 4; ++j) {
    int i = threadIdx.x + j * 256;
    ho[i] = f2bf(v[j] * scale * w[i]);
  }
}

// ---------------- generic bf16 WMMA GEMM: C[M,N] f32 = A[M,K] @ B[K,N] ----
// block tile 128x128, 8 waves (4x2), wave 32x64 via 2x4 wmma accumulators.
// A tile optionally DMA'd by the Tensor Data Mover (wave 0 issues, waits on
// TENSORcnt, block barrier publishes). N must be a multiple of 128.
__global__ __launch_bounds__(256) void gemm_bf16_kernel(
    const unsigned short* __restrict__ A, const unsigned short* __restrict__ B,
    float* __restrict__ C, int M, int N, int K)
{
  __shared__ unsigned short sA [128][32];   // [row][k]
  __shared__ unsigned short sBt[128][32];   // [col][k] (transposed tile)
  const int tid  = threadIdx.x;
  const int lane = tid & 31;
  const int wave = tid >> 5;
  const int half = lane >> 4;
  const int l15  = lane & 15;
  const int wm   = wave >> 1;   // 0..3 -> 32-row strip
  const int wn   = wave & 1;    // 0..1 -> 64-col strip
  const int bm   = blockIdx.y;
  const int bn   = blockIdx.x;

  v8f acc[2][4] = {};

  for (int k0 = 0; k0 < K; k0 += 32) {
#if HAVE_TDM
    if (wave == 0) {
      // D# for a 2D tile: 128 rows x 32 bf16, row stride K elements.
      unsigned lds_off = (unsigned)(size_t)(&sA[0][0]);
      unsigned long long ga =
          (unsigned long long)(size_t)(A + (size_t)(bm * 128) * K + k0);
      v4u g0 = {};
      g0[0] = 1u;                                            // count=1 (user D#)
      g0[1] = lds_off;                                       // lds_addr
      g0[2] = (unsigned)ga;                                  // global_addr[31:0]
      g0[3] = (unsigned)((ga >> 32) & 0x01ffffffu) | (2u << 30); // addr[56:32], type=2
      v8i g1 = {};
      g1[0] = (int)(1u << 16);                               // data_size=1 -> 2 bytes
      g1[1] = (int)(((unsigned)K & 0xffffu) << 16);          // tensor_dim0[15:0]
      g1[2] = (int)((((unsigned)K >> 16) & 0xffffu) | (128u << 16)); // dim0 hi | dim1 lo
      g1[3] = (int)(32u << 16);                              // tile_dim0 = 32
      g1[4] = 128;                                           // tile_dim1 = 128
      g1[5] = (int)(unsigned)K;                              // tensor_dim0_stride
      v4i gz4 = {};
      v8i gz8 = {};
      __builtin_amdgcn_tensor_load_to_lds(g0, g1, gz4, gz4, gz8, 0);
      __builtin_amdgcn_s_wait_tensorcnt(0);
    }
#else
    { // A tile 128x32, manual path
      int row = tid >> 1;
      int seg = (tid & 1) << 4;
      const uint4* src = (const uint4*)(A + (size_t)(bm * 128 + row) * K + k0 + seg);
      uint4* dst = (uint4*)(&sA[row][seg]);
      dst[0] = src[0];
      dst[1] = src[1];
    }
#endif
    { // B tile 32x128 -> transposed into sBt[col][k]
      int kk  = tid >> 3;        // 0..31
      int seg = (tid & 7) << 4;  // 0..112
      const unsigned short* src = B + (size_t)(k0 + kk) * N + bn * 128 + seg;
      union { uint4 v; unsigned short s[8]; } u0, u1;
      u0.v = *(const uint4*)(src);
      u1.v = *(const uint4*)(src + 8);
      #pragma unroll
      for (int j = 0; j < 8; ++j) sBt[seg + j][kk] = u0.s[j];
      #pragma unroll
      for (int j = 0; j < 8; ++j) sBt[seg + 8 + j][kk] = u1.s[j];
    }
    if (k0 + 64 < K) {  // speculative prefetch of the tile after next
      __builtin_prefetch(A + (size_t)(bm * 128 + (tid >> 1)) * K + k0 + 64, 0, 1);
      __builtin_prefetch(B + (size_t)(k0 + 64 + (tid >> 3)) * N + bn * 128, 0, 1);
    }
    __syncthreads();

    v16bf af[2], bfr[4];
    #pragma unroll
    for (int i = 0; i < 2; ++i) {
      int row = wm * 32 + i * 16 + l15;
      af[i] = ldfrag(&sA[row][8 * half], &sA[row][16 + 8 * half]);
    }
    #pragma unroll
    for (int j = 0; j < 4; ++j) {
      int col = wn * 64 + j * 16 + l15;
      bfr[j] = ldfrag(&sBt[col][8 * half], &sBt[col][16 + 8 * half]);
    }
    #pragma unroll
    for (int i = 0; i < 2; ++i)
      #pragma unroll
      for (int j = 0; j < 4; ++j)
        acc[i][j] = __builtin_amdgcn_wmma_f32_16x16x32_bf16(
            false, af[i], false, bfr[j], (short)0, acc[i][j], false, false);
    __syncthreads();
  }

  #pragma unroll
  for (int i = 0; i < 2; ++i)
    #pragma unroll
    for (int j = 0; j < 4; ++j) {
      int gr0 = bm * 128 + wm * 32 + i * 16 + half * 8;  // f32 C/D layout
      int gc  = bn * 128 + wn * 64 + j * 16 + l15;
      #pragma unroll
      for (int r = 0; r < 8; ++r)
        C[(size_t)(gr0 + r) * N + gc] = acc[i][j][r];
    }
}

// ---------------- depthwise causal conv + SiLU + dt softplus ----------------
__global__ void convdt_kernel(const float* __restrict__ zx,
    const float* __restrict__ cw, const float* __restrict__ cb,
    const float* __restrict__ dt_bias,
    unsigned short* __restrict__ xq, unsigned short* __restrict__ Bq,
    unsigned short* __restrict__ Cq, float* __restrict__ dt)
{
  const int PER = D_CONV_CH + NHEADS;   // 2336
  long long idx = (long long)blockIdx.x * 256 + threadIdx.x;
  long long total = (long long)ROWS * PER;
  if (idx >= total) return;
  int c   = (int)(idx % PER);
  int row = (int)(idx / PER);
  int l   = row % SEQLEN;
  if (c < D_CONV_CH) {
    float acc = cb[c];
    #pragma unroll
    for (int w = 0; w < D_CONV; ++w) {
      int ls = l - (D_CONV - 1) + w;
      if (ls >= 0)
        acc += cw[w * D_CONV_CH + c] *
               zx[(size_t)(row + ls - l) * ZXS + D_INNER + c];
    }
    acc = acc * sigmoidf_(acc);
    unsigned short q = f2bf(acc);
    if (c < D_INNER)
      xq[(size_t)row * D_INNER + c] = q;
    else if (c < D_INNER + D_STATE)
      Bq[(size_t)row * D_STATE + (c - D_INNER)] = q;
    else
      Cq[(size_t)row * D_STATE + (c - D_INNER - D_STATE)] = q;
  } else {
    int h = c - D_CONV_CH;
    float x = zx[(size_t)row * ZXS + (2 * D_INNER + 2 * D_STATE) + h] + dt_bias[h];
    dt[(size_t)row * NHEADS + h] = (x > 20.f) ? x : log1pf(__expf(x));
  }
}

// ---------------- per (b,h,chunk) cumsum of dt*A ----------------
__global__ __launch_bounds__(256) void acum_kernel(
    const float* __restrict__ dt, const float* __restrict__ A_log,
    float* __restrict__ acum)
{
  int h = blockIdx.x, cb = blockIdx.y, b = blockIdx.z;
  int l = threadIdx.x;
  int row = b * SEQLEN + cb * CHUNK + l;
  float Ah = -__expf(A_log[h]);
  __shared__ float s[256];
  s[l] = dt[(size_t)row * NHEADS + h] * Ah;
  __syncthreads();
  for (int off = 1; off < 256; off <<= 1) {
    float t = (l >= off) ? s[l - off] : 0.f;
    __syncthreads();
    s[l] += t;
    __syncthreads();
  }
  acum[((size_t)(b * NHEADS + h) * NCHUNK + cb) * CHUNK + l] = s[l];
}

// ---------------- G = C * B^T per (b,chunk): WMMA, 128x128 tile / block ----
__global__ __launch_bounds__(256) void gmat_kernel(
    const unsigned short* __restrict__ Bq, const unsigned short* __restrict__ Cq,
    float* __restrict__ G)
{
  __shared__ unsigned short sC[128][32];   // [l][n]
  __shared__ unsigned short sS[128][32];   // [s][n] == [col][k]
  const int tid  = threadIdx.x;
  const int lane = tid & 31;
  const int wave = tid >> 5;
  const int half = lane >> 4;
  const int l15  = lane & 15;
  const int wm   = wave >> 1;   // 0..3 -> 32 l-rows
  const int wn   = wave & 1;    // 0..1 -> 64 s-cols
  const int bm   = blockIdx.y, bn = blockIdx.x;
  const int bc   = blockIdx.z;
  const int b = bc / NCHUNK, cb = bc % NCHUNK;
  const int row0 = b * SEQLEN + cb * CHUNK;

  v8f acc[2][4] = {};

  for (int n0 = 0; n0 < D_STATE; n0 += 32) {
    {
      int row = tid >> 1;
      int seg = (tid & 1) << 4;
      const uint4* srcC = (const uint4*)(Cq + (size_t)(row0 + bm * 128 + row) * D_STATE + n0 + seg);
      uint4* dC = (uint4*)(&sC[row][seg]);
      dC[0] = srcC[0]; dC[1] = srcC[1];
      const uint4* srcB = (const uint4*)(Bq + (size_t)(row0 + bn * 128 + row) * D_STATE + n0 + seg);
      uint4* dB = (uint4*)(&sS[row][seg]);
      dB[0] = srcB[0]; dB[1] = srcB[1];
    }
    __syncthreads();
    v16bf af[2], bfr[4];
    #pragma unroll
    for (int i = 0; i < 2; ++i) {
      int row = wm * 32 + i * 16 + l15;
      af[i] = ldfrag(&sC[row][8 * half], &sC[row][16 + 8 * half]);
    }
    #pragma unroll
    for (int j = 0; j < 4; ++j) {
      int col = wn * 64 + j * 16 + l15;
      bfr[j] = ldfrag(&sS[col][8 * half], &sS[col][16 + 8 * half]);
    }
    #pragma unroll
    for (int i = 0; i < 2; ++i)
      #pragma unroll
      for (int j = 0; j < 4; ++j)
        acc[i][j] = __builtin_amdgcn_wmma_f32_16x16x32_bf16(
            false, af[i], false, bfr[j], (short)0, acc[i][j], false, false);
    __syncthreads();
  }

  float* Gp = G + (size_t)bc * CHUNK * CHUNK;
  #pragma unroll
  for (int i = 0; i < 2; ++i)
    #pragma unroll
    for (int j = 0; j < 4; ++j) {
      int gl = bm * 128 + wm * 32 + i * 16 + half * 8;
      int gs = bn * 128 + wn * 64 + j * 16 + l15;
      #pragma unroll
      for (int r = 0; r < 8; ++r)
        Gp[(size_t)(gl + r) * CHUNK + gs] = acc[i][j][r];
    }
}

// ---------------- chunk end-states via WMMA -------------------------------
// states[p][n] = sum_l (x[l][p]*decay[l]*dt[l]) * B[l][n]   (M=64,N=128,K=256)
__global__ __launch_bounds__(256) void states_kernel(
    const unsigned short* __restrict__ xq, const unsigned short* __restrict__ Bq,
    const float* __restrict__ dt, const float* __restrict__ acum,
    float* __restrict__ states)
{
  int h = blockIdx.x, cb = blockIdx.y, b = blockIdx.z;
  extern __shared__ unsigned char ssm[];
  unsigned short* sxw = (unsigned short*)ssm;      // [p 64][l 256] scaled-x, A-op
  unsigned short* sBt = sxw + 64 * 256;            // [n 128][l 256] B-op
  const int tid  = threadIdx.x;
  const int lane = tid & 31;
  const int wave = tid >> 5;
  const int half = lane >> 4;
  const int l15  = lane & 15;
  const int wm   = wave >> 2;   // 0..1 -> 32 p-rows
  const int wn   = wave & 3;    // 0..3 -> 32 n-cols
  const int row0 = b * SEQLEN + cb * CHUNK;
  const float* ac = acum + ((size_t)(b * NHEADS + h) * NCHUNK + cb) * CHUNK;

  { // fill: thread owns sequence position l = tid
    int l = tid;
    float wl = __expf(ac[CHUNK - 1] - ac[l]) * dt[(size_t)(row0 + l) * NHEADS + h];
    const unsigned short* xrow = xq + (size_t)(row0 + l) * D_INNER + h * HEADDIM;
    #pragma unroll
    for (int q = 0; q < 8; ++q) {
      union { uint4 v; unsigned short s[8]; } u;
      u.v = *(const uint4*)(xrow + q * 8);
      #pragma unroll
      for (int j = 0; j < 8; ++j)
        sxw[(q * 8 + j) * 256 + l] = f2bf(bf2f(u.s[j]) * wl);
    }
    const unsigned short* brow = Bq + (size_t)(row0 + l) * D_STATE;
    #pragma unroll
    for (int q = 0; q < 16; ++q) {
      union { uint4 v; unsigned short s[8]; } u;
      u.v = *(const uint4*)(brow + q * 8);
      #pragma unroll
      for (int j = 0; j < 8; ++j)
        sBt[(q * 8 + j) * 256 + l] = u.s[j];
    }
  }
  __syncthreads();

  v8f acc[2][2] = {};
  for (int k0 = 0; k0 < CHUNK; k0 += 32) {
    v16bf af[2], bfr[2];
    #pragma unroll
    for (int i = 0; i < 2; ++i) {
      const unsigned short* base = sxw + (wm * 32 + i * 16 + l15) * 256 + k0;
      af[i] = ldfrag(base + 8 * half, base + 16 + 8 * half);
    }
    #pragma unroll
    for (int j = 0; j < 2; ++j) {
      const unsigned short* base = sBt + (wn * 32 + j * 16 + l15) * 256 + k0;
      bfr[j] = ldfrag(base + 8 * half, base + 16 + 8 * half);
    }
    #pragma unroll
    for (int i = 0; i < 2; ++i)
      #pragma unroll
      for (int j = 0; j < 2; ++j)
        acc[i][j] = __builtin_amdgcn_wmma_f32_16x16x32_bf16(
            false, af[i], false, bfr[j], (short)0, acc[i][j], false, false);
  }

  float* sp = states + ((size_t)(b * NCHUNK + cb) * NHEADS + h) * (HEADDIM * D_STATE);
  #pragma unroll
  for (int i = 0; i < 2; ++i)
    #pragma unroll
    for (int j = 0; j < 2; ++j) {
      int p0 = wm * 32 + i * 16 + half * 8;
      int n  = wn * 32 + j * 16 + l15;
      #pragma unroll
      for (int r = 0; r < 8; ++r)
        sp[(size_t)(p0 + r) * D_STATE + n] = acc[i][j][r];
    }
}

// ---------------- sequential inter-chunk scan (16 steps) ----------------
__global__ __launch_bounds__(256) void scan_kernel(
    const float* __restrict__ states, const float* __restrict__ acum,
    float* __restrict__ prev)
{
  int h = blockIdx.x, b = blockIdx.y;
  int tid = threadIdx.x;
  float carry[32];
  #pragma unroll
  for (int j = 0; j < 32; ++j) carry[j] = 0.f;
  for (int c = 0; c < NCHUNK; ++c) {
    float dec = __expf(acum[((size_t)(b * NHEADS + h) * NCHUNK + c) * CHUNK + CHUNK - 1]);
    size_t base = ((size_t)(b * NCHUNK + c) * NHEADS + h) * (HEADDIM * D_STATE);
    for (int j = 0; j < 32; ++j) {
      size_t o = base + (size_t)tid * 32 + j;
      prev[o] = carry[j];                       // state entering chunk c
      carry[j] = carry[j] * dec + states[o];
    }
  }
}

// ---------------- Y via WMMA: intra (causal) + inter + D-skip --------------
__global__ __launch_bounds__(256) void ydiag_kernel(
    const unsigned short* __restrict__ xq, const unsigned short* __restrict__ Cq,
    const float* __restrict__ G, const float* __restrict__ prev,
    const float* __restrict__ dt, const float* __restrict__ acum,
    const float* __restrict__ Dp, float* __restrict__ y)
{
  int h = blockIdx.x, cb = blockIdx.y, b = blockIdx.z;
  extern __shared__ unsigned char ysm[];
  unsigned short* sxT = (unsigned short*)ysm;        // [p 64][s 256] x, B-op
  unsigned short* sW  = sxT + 64 * 256;              // [wave 8][l 32][s 32] A-op
  unsigned short* sCa = sW + 8 * 32 * 32;            // [l 256][n 128] C*eAl, A-op
  unsigned short* sPv = sCa + 256 * 128;             // [p 64][n 128] prev, B-op
  float* sac = (float*)(sPv + 64 * 128);             // 256
  float* sdt = sac + 256;                            // 256
  const int tid  = threadIdx.x;
  const int lane = tid & 31;
  const int wv   = tid >> 5;
  const int half = lane >> 4;
  const int l15  = lane & 15;
  const int row0 = b * SEQLEN + cb * CHUNK;

  { // fill: thread owns sequence position l = tid
    int l = tid;
    float al = acum[((size_t)(b * NHEADS + h) * NCHUNK + cb) * CHUNK + l];
    sac[l] = al;
    sdt[l] = dt[(size_t)(row0 + l) * NHEADS + h];
    float eAl = __expf(al);
    const unsigned short* xrow = xq + (size_t)(row0 + l) * D_INNER + h * HEADDIM;
    #pragma unroll
    for (int q = 0; q < 8; ++q) {
      union { uint4 v; unsigned short s[8]; } u;
      u.v = *(const uint4*)(xrow + q * 8);
      #pragma unroll
      for (int j = 0; j < 8; ++j)
        sxT[(q * 8 + j) * 256 + l] = u.s[j];
    }
    const unsigned short* crow = Cq + (size_t)(row0 + l) * D_STATE;
    #pragma unroll
    for (int q = 0; q < 16; ++q) {
      union { uint4 v; unsigned short s[8]; } ui, uo;
      ui.v = *(const uint4*)(crow + q * 8);
      #pragma unroll
      for (int j = 0; j < 8; ++j) uo.s[j] = f2bf(bf2f(ui.s[j]) * eAl);
      *(uint4*)(sCa + l * 128 + q * 8) = uo.v;
    }
    size_t pb = ((size_t)(b * NCHUNK + cb) * NHEADS + h) * (HEADDIM * D_STATE);
    for (int j = 0; j < 32; ++j) {
      int idx = tid * 32 + j;
      sPv[idx] = f2bf(prev[pb + idx]);
    }
  }
  __syncthreads();

  v8f acc[2][4] = {};
  const float* Gp = G + (size_t)(b * NCHUNK + cb) * CHUNK * CHUNK;

  // ---- intra-chunk (causal over s-tiles; trip count uniform per wave) ----
  unsigned short* myW = sW + wv * 32 * 32;
  int lg = wv * 32 + lane;           // this lane's l-row for W building
  float Al = sac[lg];
  for (int st = 0; st <= wv; ++st) {
    #pragma unroll 8
    for (int ss = 0; ss < 32; ++ss) {
      int sg = st * 32 + ss;
      float v = 0.f;
      if (sg <= lg)
        v = __expf(Al - sac[sg]) * Gp[(size_t)lg * CHUNK + sg] * sdt[sg];
      myW[lane * 32 + ss] = f2bf(v);
    }
    __builtin_amdgcn_wave_barrier();   // keep LDS writes before frag reads
    v16bf af[2], bfr[4];
    #pragma unroll
    for (int i = 0; i < 2; ++i) {
      const unsigned short* base = myW + (i * 16 + l15) * 32;
      af[i] = ldfrag(base + 8 * half, base + 16 + 8 * half);
    }
    #pragma unroll
    for (int j = 0; j < 4; ++j) {
      const unsigned short* base = sxT + (j * 16 + l15) * 256 + st * 32;
      bfr[j] = ldfrag(base + 8 * half, base + 16 + 8 * half);
    }
    #pragma unroll
    for (int i = 0; i < 2; ++i)
      #pragma unroll
      for (int j = 0; j < 4; ++j)
        acc[i][j] = __builtin_amdgcn_wmma_f32_16x16x32_bf16(
            false, af[i], false, bfr[j], (short)0, acc[i][j], false, false);
    __builtin_amdgcn_wave_barrier();
  }

  // ---- inter-chunk: (C*eAl) @ prev^T, K = 128 ----
  for (int k0 = 0; k0 < D_STATE; k0 += 32) {
    v16bf af[2], bfr[4];
    #pragma unroll
    for (int i = 0; i < 2; ++i) {
      const unsigned short* base = sCa + (wv * 32 + i * 16 + l15) * 128 + k0;
      af[i] = ldfrag(base + 8 * half, base + 16 + 8 * half);
    }
    #pragma unroll
    for (int j = 0; j < 4; ++j) {
      const unsigned short* base = sPv + (j * 16 + l15) * 128 + k0;
      bfr[j] = ldfrag(base + 8 * half, base + 16 + 8 * half);
    }
    #pragma unroll
    for (int i = 0; i < 2; ++i)
      #pragma unroll
      for (int j = 0; j < 4; ++j)
        acc[i][j] = __builtin_amdgcn_wmma_f32_16x16x32_bf16(
            false, af[i], false, bfr[j], (short)0, acc[i][j], false, false);
  }

  // ---- D-skip + store ----
  float Dh = Dp[h];
  #pragma unroll
  for (int i = 0; i < 2; ++i)
    #pragma unroll
    for (int j = 0; j < 4; ++j) {
      int l0 = wv * 32 + i * 16 + half * 8;
      int p  = j * 16 + l15;
      #pragma unroll
      for (int r = 0; r < 8; ++r) {
        float xv = bf2f(sxT[p * 256 + l0 + r]);
        y[(size_t)(row0 + l0 + r) * D_INNER + h * HEADDIM + p] = acc[i][j][r] + Dh * xv;
      }
    }
}

// ---------------- gated RMSNorm -> bf16 ----------------
__global__ __launch_bounds__(256) void gate_kernel(
    const float* __restrict__ y, const float* __restrict__ zx,
    const float* __restrict__ gw, unsigned short* __restrict__ yq)
{
  int row = blockIdx.x;
  const float* yp = y + (size_t)row * D_INNER;
  const float* zp = zx + (size_t)row * ZXS;   // z = channels [0, 2048)
  __shared__ float red[256];
  float g[8]; float ss = 0.f;
  #pragma unroll
  for (int j = 0; j < 8; ++j) {
    int i = threadIdx.x + j * 256;
    float z = zp[i];
    float v = yp[i] * z * sigmoidf_(z);
    g[j] = v; ss += v * v;
  }
  red[threadIdx.x] = ss; __syncthreads();
  for (int o = 128; o > 0; o >>= 1) {
    if ((int)threadIdx.x < o) red[threadIdx.x] += red[threadIdx.x + o];
    __syncthreads();
  }
  float scale = rsqrtf(red[0] / (float)D_INNER + EPS);
  unsigned short* yo = yq + (size_t)row * D_INNER;
  #pragma unroll
  for (int j = 0; j < 8; ++j) {
    int i = threadIdx.x + j * 256;
    yo[i] = f2bf(g[j] * scale * gw[i]);
  }
}

// ---------------- launch ----------------
extern "C" void kernel_launch(void* const* d_in, const int* in_sizes, int n_in,
                              void* d_out, int out_size, void* d_ws, size_t ws_size,
                              hipStream_t stream) {
  (void)in_sizes; (void)n_in; (void)out_size; (void)ws_size;
  const float* hidden     = (const float*)d_in[0];
  const float* residual   = (const float*)d_in[1];
  const float* norm_w     = (const float*)d_in[2];
  const float* in_proj_w  = (const float*)d_in[3];
  const float* conv_w     = (const float*)d_in[4];
  const float* conv_b     = (const float*)d_in[5];
  const float* dt_bias    = (const float*)d_in[6];
  const float* A_log      = (const float*)d_in[7];
  const float* Dp         = (const float*)d_in[8];
  const float* gate_w     = (const float*)d_in[9];
  const float* out_proj_w = (const float*)d_in[10];

  float* out     = (float*)d_out;
  float* res_out = out + (size_t)ROWS * D_MODEL;

  char* w = (char*)d_ws;
  size_t off = 0;
  auto alloc = [&](size_t bytes) { size_t o = off; off += (bytes + 255) & ~(size_t)255; return o; };
  unsigned short* h_bf   = (unsigned short*)(w + alloc((size_t)ROWS * D_MODEL * 2));
  unsigned short* win_bf = (unsigned short*)(w + alloc((size_t)D_MODEL * ZXS * 2));
  unsigned short* wout_bf= (unsigned short*)(w + alloc((size_t)D_INNER * D_MODEL * 2));
  float*          zx     = (float*)        (w + alloc((size_t)ROWS * ZXS * 4));
  unsigned short* xq     = (unsigned short*)(w + alloc((size_t)ROWS * D_INNER * 2));
  unsigned short* Bq     = (unsigned short*)(w + alloc((size_t)ROWS * D_STATE * 2));
  unsigned short* Cq     = (unsigned short*)(w + alloc((size_t)ROWS * D_STATE * 2));
  float*          dt     = (float*)        (w + alloc((size_t)ROWS * NHEADS * 4));
  float*          acum   = (float*)        (w + alloc((size_t)BATCH * NHEADS * NCHUNK * CHUNK * 4));
  float*          G      = (float*)        (w + alloc((size_t)BATCH * NCHUNK * CHUNK * CHUNK * 4));
  float*          states = (float*)        (w + alloc((size_t)BATCH * NCHUNK * NHEADS * HEADDIM * D_STATE * 4));
  float*          prev   = (float*)        (w + alloc((size_t)BATCH * NCHUNK * NHEADS * HEADDIM * D_STATE * 4));
  float*          yfull  = (float*)        (w + alloc((size_t)ROWS * D_INNER * 4));
  unsigned short* yq     = (unsigned short*)(w + alloc((size_t)ROWS * D_INNER * 2));

  // 1. weight conversion (in_proj zero-padded to ZXS columns)
  {
    long long nWin = (long long)D_MODEL * ZXS;
    f2bf_pad_kernel<<<(unsigned)((nWin + 255) / 256), 256, 0, stream>>>(
        in_proj_w, win_bf, D_IN_PROJ, ZXS, nWin);
    long long nWout = (long long)D_INNER * D_MODEL;
    f2bf_kernel<<<(unsigned)((nWout + 255) / 256), 256, 0, stream>>>(out_proj_w, wout_bf, nWout);
  }
  // 2. residual + RMSNorm
  addnorm_kernel<<<ROWS, 256, 0, stream>>>(hidden, residual, norm_w, res_out, h_bf);
  // 3. in_proj GEMM: [8192,1024] x [1024,4480(padded)]
  gemm_bf16_kernel<<<dim3(ZXS / 128, ROWS / 128), 256, 0, stream>>>(
      h_bf, win_bf, zx, ROWS, ZXS, D_MODEL);
  // 4. conv + SiLU + dt softplus
  {
    long long total = (long long)ROWS * (D_CONV_CH + NHEADS);
    convdt_kernel<<<(unsigned)((total + 255) / 256), 256, 0, stream>>>(
        zx, conv_w, conv_b, dt_bias, xq, Bq, Cq, dt);
  }
  // 5. cumulative dt*A
  acum_kernel<<<dim3(NHEADS, NCHUNK, BATCH), 256, 0, stream>>>(dt, A_log, acum);
  // 6. G = C B^T per (b,chunk), WMMA
  gmat_kernel<<<dim3(2, 2, BATCH * NCHUNK), 256, 0, stream>>>(Bq, Cq, G);
  // 7. chunk end-states, WMMA
  states_kernel<<<dim3(NHEADS, NCHUNK, BATCH), 256,
                  (64 * 256 + 128 * 256) * sizeof(unsigned short),
                  stream>>>(xq, Bq, dt, acum, states);
  // 8. inter-chunk recurrence
  scan_kernel<<<dim3(NHEADS, BATCH), 256, 0, stream>>>(states, acum, prev);
  // 9. Y combine (intra + inter + D skip), WMMA
  ydiag_kernel<<<dim3(NHEADS, NCHUNK, BATCH), 256,
                 (64 * 256 + 8 * 32 * 32 + 256 * 128 + 64 * 128) * sizeof(unsigned short)
                 + 2 * 256 * sizeof(float),
                 stream>>>(xq, Cq, G, prev, dt, acum, Dp, yfull);
  // 10. gated RMSNorm -> bf16
  gate_kernel<<<ROWS, 256, 0, stream>>>(yfull, zx, gate_w, yq);
  // 11. out_proj GEMM: [8192,2048] x [2048,1024] -> out
  gemm_bf16_kernel<<<dim3(D_MODEL / 128, ROWS / 128), 256, 0, stream>>>(
      yq, wout_bf, out, ROWS, D_MODEL, D_INNER);
}
